// MGraphDTA_Hetero_51058571215013
// MI455X (gfx1250) — matrix operand order repaired and checked by reading, not compiled, not run on hardware
//
#include <hip/hip_runtime.h>
#include <math.h>

#define H 128
#define ACT_NONE 0
#define ACT_RELU 1
#define ACT_GELU 2

typedef __attribute__((ext_vector_type(16))) _Float16 v16h;
typedef __attribute__((ext_vector_type(8)))  float    v8f;

static inline int cdiv64(long long a, int b) { return (int)((a + b - 1) / b); }

// ---------------- wave helpers (wave32) ----------------
__device__ __forceinline__ float wave_sum(float v) {
#pragma unroll
    for (int o = 16; o > 0; o >>= 1) v += __shfl_xor(v, o, 32);
    return v;
}

__device__ __forceinline__ void atomic_max_f32(float* addr, float val) {
    unsigned int* ua = (unsigned int*)addr;
    unsigned int old = *ua;
    while (__uint_as_float(old) < val) {
        unsigned int assumed = old;
        old = atomicCAS(ua, assumed, __float_as_uint(val));
        if (old == assumed) break;
    }
}

__device__ __forceinline__ float act_apply(float x, int ACT) {
    if (ACT == ACT_RELU) return fmaxf(x, 0.f);
    if (ACT == ACT_GELU) return 0.5f * x * (1.f + erff(x * 0.70710678118f));
    return x;
}

// ---------------- WMMA GEMM fast path ----------------
// C = act(A[NxK] @ W[KxO] + bias). Requires N%16==0, K%32==0, O%32==0.
// One wave computes a 16x32 C strip (two 16x16 WMMA tiles sharing the A fragment).
// A fragment: four 16B global_load_b128; B fragments: unpredicated b32 loads.
template<int ACT>
__global__ void gemm_wmma_fast_k(const float* __restrict__ A, const float* __restrict__ W,
                                 const float* __restrict__ bias, float* __restrict__ C,
                                 int N, int K, int O)
{
    int wid  = blockIdx.x * (blockDim.x >> 5) + (threadIdx.x >> 5);
    int lane = threadIdx.x & 31;
    int gO = O >> 5;                 // 32-wide column groups
    int gN = N >> 4;
    if (wid >= gN * gO) return;      // wave-uniform: EXEC stays all-1s
    int row0 = (wid / gO) << 4;
    int col0 = (wid % gO) << 5;
    int r = lane & 15, hi = lane >> 4;
    const float* arowp = A + (size_t)(row0 + r) * K;
    int bcol0 = col0 + r;
    int bcol1 = col0 + 16 + r;
    v8f acc0 = {0.f,0.f,0.f,0.f,0.f,0.f,0.f,0.f};
    v8f acc1 = {0.f,0.f,0.f,0.f,0.f,0.f,0.f,0.f};
    for (int k0 = 0; k0 < K; k0 += 32) {
        // A 16x32 f16 fragment (ISA 7.12.2): lane half selects 8-wide K groups
        const float4* ap0 = (const float4*)(arowp + k0 + hi * 8);
        const float4* ap1 = (const float4*)(arowp + k0 + 16 + hi * 8);
        float4 x0 = ap0[0], x1 = ap0[1], x2 = ap1[0], x3 = ap1[1];
        __builtin_prefetch(arowp + k0 + 32, 0, 0);
        v16h a;
        a[0]  = (_Float16)x0.x; a[1]  = (_Float16)x0.y; a[2]  = (_Float16)x0.z; a[3]  = (_Float16)x0.w;
        a[4]  = (_Float16)x1.x; a[5]  = (_Float16)x1.y; a[6]  = (_Float16)x1.z; a[7]  = (_Float16)x1.w;
        a[8]  = (_Float16)x2.x; a[9]  = (_Float16)x2.y; a[10] = (_Float16)x2.z; a[11] = (_Float16)x2.w;
        a[12] = (_Float16)x3.x; a[13] = (_Float16)x3.y; a[14] = (_Float16)x3.z; a[15] = (_Float16)x3.w;
        // B 32x16 fragments: col = lane&15, lanes 0-15 K=0..7,16..23, lanes 16-31 K=8..15,24..31
        v16h b0, b1;
#pragma unroll
        for (int j = 0; j < 8; ++j) {
            int kk = k0 + ((j < 4) ? (hi * 8 + 2 * j) : (16 + hi * 8 + 2 * (j - 4)));
            const float* w0 = W + (size_t)kk * O;
            const float* w1 = W + (size_t)(kk + 1) * O;
            b0[2 * j]     = (_Float16)w0[bcol0];
            b0[2 * j + 1] = (_Float16)w1[bcol0];
            b1[2 * j]     = (_Float16)w0[bcol1];
            b1[2 * j + 1] = (_Float16)w1[bcol1];
        }
        acc0 = __builtin_amdgcn_wmma_f32_16x16x32_f16(false, a, false, b0, (short)0, acc0, false, false);
        acc1 = __builtin_amdgcn_wmma_f32_16x16x32_f16(false, a, false, b1, (short)0, acc1, false, false);
    }
    float bv0 = bias ? bias[bcol0] : 0.f;
    float bv1 = bias ? bias[bcol1] : 0.f;
#pragma unroll
    for (int v = 0; v < 8; ++v) {
        int rowv = row0 + v + (hi << 3);          // C/D: M = vgpr + 8*(lane>=16)
        C[(size_t)rowv * O + bcol0] = act_apply(acc0[v] + bv0, ACT);
        C[(size_t)rowv * O + bcol1] = act_apply(acc1[v] + bv1, ACT);
    }
}

// ---------------- WMMA GEMM padded-K path (projections; K arbitrary) ----------------
// Same tiling; K guarded via clamp-index + select (v_cndmask, no exec branches).
template<int ACT>
__global__ void gemm_wmma_pad_k(const float* __restrict__ A, const float* __restrict__ W,
                                const float* __restrict__ bias, float* __restrict__ C,
                                int N, int K, int O)
{
    int wid  = blockIdx.x * (blockDim.x >> 5) + (threadIdx.x >> 5);
    int lane = threadIdx.x & 31;
    int gO = O >> 5;
    int gN = N >> 4;
    if (wid >= gN * gO) return;
    int row0 = (wid / gO) << 4;
    int col0 = (wid % gO) << 5;
    int r = lane & 15, hi = lane >> 4;
    const float* arowp = A + (size_t)(row0 + r) * K;
    int bcol0 = col0 + r;
    int bcol1 = col0 + 16 + r;
    v8f acc0 = {0.f,0.f,0.f,0.f,0.f,0.f,0.f,0.f};
    v8f acc1 = {0.f,0.f,0.f,0.f,0.f,0.f,0.f,0.f};
    for (int k0 = 0; k0 < K; k0 += 32) {
        v16h a, b0, b1;
#pragma unroll
        for (int j = 0; j < 8; ++j) {
            int kk = k0 + ((j < 4) ? (hi * 8 + 2 * j) : (16 + hi * 8 + 2 * (j - 4)));
            int k_0 = (kk     < K) ? kk     : 0;
            int k_1 = (kk + 1 < K) ? kk + 1 : 0;
            float x0 = arowp[k_0]; x0 = (kk     < K) ? x0 : 0.f;
            float x1 = arowp[k_1]; x1 = (kk + 1 < K) ? x1 : 0.f;
            a[2 * j]     = (_Float16)x0;
            a[2 * j + 1] = (_Float16)x1;
        }
#pragma unroll
        for (int j = 0; j < 8; ++j) {
            int kk = k0 + ((j < 4) ? (hi * 8 + 2 * j) : (16 + hi * 8 + 2 * (j - 4)));
            int k_0 = (kk     < K) ? kk     : 0;
            int k_1 = (kk + 1 < K) ? kk + 1 : 0;
            float w00 = W[(size_t)k_0 * O + bcol0]; w00 = (kk     < K) ? w00 : 0.f;
            float w10 = W[(size_t)k_1 * O + bcol0]; w10 = (kk + 1 < K) ? w10 : 0.f;
            float w01 = W[(size_t)k_0 * O + bcol1]; w01 = (kk     < K) ? w01 : 0.f;
            float w11 = W[(size_t)k_1 * O + bcol1]; w11 = (kk + 1 < K) ? w11 : 0.f;
            b0[2 * j] = (_Float16)w00; b0[2 * j + 1] = (_Float16)w10;
            b1[2 * j] = (_Float16)w01; b1[2 * j + 1] = (_Float16)w11;
        }
        acc0 = __builtin_amdgcn_wmma_f32_16x16x32_f16(false, a, false, b0, (short)0, acc0, false, false);
        acc1 = __builtin_amdgcn_wmma_f32_16x16x32_f16(false, a, false, b1, (short)0, acc1, false, false);
    }
    float bv0 = bias ? bias[bcol0] : 0.f;
    float bv1 = bias ? bias[bcol1] : 0.f;
#pragma unroll
    for (int v = 0; v < 8; ++v) {
        int rowv = row0 + v + (hi << 3);
        C[(size_t)rowv * O + bcol0] = act_apply(acc0[v] + bv0, ACT);
        C[(size_t)rowv * O + bcol1] = act_apply(acc1[v] + bv1, ACT);
    }
}

// ---------------- batch-norm (training-mode batch stats) ----------------
__global__ void col_meanvar_k(const float* __restrict__ X, int N, float* mu, float* var) {
    __shared__ float sh[256], sh2[256];
    int c = blockIdx.x;
    float s = 0.f, s2 = 0.f;
    for (int i = threadIdx.x; i < N; i += blockDim.x) {
        float x = X[(size_t)i * H + c];
        s += x; s2 += x * x;
    }
    sh[threadIdx.x] = s; sh2[threadIdx.x] = s2;
    __syncthreads();
    for (int o = 128; o > 0; o >>= 1) {
        if ((int)threadIdx.x < o) { sh[threadIdx.x] += sh[threadIdx.x + o]; sh2[threadIdx.x] += sh2[threadIdx.x + o]; }
        __syncthreads();
    }
    if (threadIdx.x == 0) {
        float m = sh[0] / (float)N;
        mu[c] = m; var[c] = sh2[0] / (float)N - m * m;
    }
}

__global__ void bn_apply_k(const float* x, const float* mu, const float* var,
                           const float* g, const float* b, float* y, long long tot) {
    long long t = blockIdx.x * (long long)blockDim.x + threadIdx.x;
    if (t >= tot) return;
    int c = (int)(t & (H - 1));
    y[t] = (x[t] - mu[c]) * rsqrtf(var[c] + 1e-5f) * g[c] + b[c];
}

// ---------------- fills / elementwise ----------------
__global__ void fill_f32_k(float* p, float v, long long n) {
    long long i = blockIdx.x * (long long)blockDim.x + threadIdx.x;
    if (i < n) p[i] = v;
}
__global__ void fill_i32_k(int* p, int v, long long n) {
    long long i = blockIdx.x * (long long)blockDim.x + threadIdx.x;
    if (i < n) p[i] = v;
}
__global__ void add2_k(const float* a, const float* b, float* o, long long n) {
    long long i = blockIdx.x * (long long)blockDim.x + threadIdx.x;
    if (i < n) o[i] = a[i] + b[i];
}
__global__ void avg2_k(const float* a, const float* b, float* o, long long n) {
    long long i = blockIdx.x * (long long)blockDim.x + threadIdx.x;
    if (i < n) o[i] = 0.5f * (a[i] + b[i]);
}
__global__ void add3_bias_k(const float* x, const float* g, const float* bias, float* z, long long n) {
    long long i = blockIdx.x * (long long)blockDim.x + threadIdx.x;
    if (i < n) { int c = (int)(i & (H - 1)); z[i] = x[i] + g[i] + bias[c]; }
}
__global__ void gin_combine_k(const float* xd, const float* agg, const float* eps, float* h, long long n) {
    long long i = blockIdx.x * (long long)blockDim.x + threadIdx.x;
    if (i < n) h[i] = (1.f + eps[0]) * xd[i] + agg[i];
}

// ---------------- segment ops ----------------
__global__ void seg_accum_ea_k(const float* ea, const int* dst, int E, int edim, float* sum, float* cnt) {
    long long t = blockIdx.x * (long long)blockDim.x + threadIdx.x;
    if (t >= (long long)E * edim) return;
    int e = (int)(t / edim), k = (int)(t % edim);
    atomicAdd(&sum[(size_t)dst[e] * edim + k], ea[t]);
    if (k == 0) atomicAdd(&cnt[dst[e]], 1.f);
}
__global__ void seg_div_k(float* sum, const float* cnt, int n, int edim) {
    long long t = blockIdx.x * (long long)blockDim.x + threadIdx.x;
    if (t >= (long long)n * edim) return;
    sum[t] /= fmaxf(cnt[t / edim], 1.f);
}
__global__ void seg_sum_rows_k(const float* xsrc, const int* src, const int* dst, int E, float* agg) {
    long long t = blockIdx.x * (long long)blockDim.x + threadIdx.x;
    if (t >= (long long)E * H) return;
    int e = (int)(t >> 7), c = (int)(t & (H - 1));
    atomicAdd(&agg[(size_t)dst[e] * H + c], xsrc[(size_t)src[e] * H + c]);
}
__global__ void batch_offsets_k(const int* batch, int n, int* offv, float* cnt) {
    int i = blockIdx.x * blockDim.x + threadIdx.x;
    if (i >= n) return;
    atomicMin(&offv[batch[i]], i);
    atomicAdd(&cnt[batch[i]], 1.f);
}
__global__ void seg_mean_pool_k(const float* x, const int* batch, const float* cnt, long long tot, float* outp) {
    long long t = blockIdx.x * (long long)blockDim.x + threadIdx.x;
    if (t >= tot) return;
    int i = (int)(t >> 7), c = (int)(t & (H - 1));
    atomicAdd(&outp[(size_t)batch[i] * H + c], x[t] / fmaxf(cnt[batch[i]], 1.f));
}

// ---------------- GATv2 edge passes (one wave per edge; lane l covers dim h*32+l) ----------------
__global__ void gat_edge_score_k(const float* xl, const float* xr, const float* ea, const float* la,
                                 const float* We, const float* att, const int* src, const int* dst,
                                 int E, int n, int edim, float* score, float* maxb)
{
    int e = blockIdx.x * (blockDim.x >> 5) + (threadIdx.x >> 5);
    int lane = threadIdx.x & 31;
    int ET = E + n;
    if (e >= ET) return;
    int s_, d_; const float* eav;
    if (e < E) { s_ = src[e]; d_ = dst[e]; eav = ea + (size_t)e * edim; }
    else       { s_ = d_ = e - E;          eav = la + (size_t)(e - E) * edim; }
    float eatt[8];
    for (int k = 0; k < edim; ++k) eatt[k] = eav[k];
#pragma unroll
    for (int hH = 0; hH < 4; ++hH) {
        int j = hH * 32 + lane;
        float ee = 0.f;
        for (int k = 0; k < edim; ++k) ee += eatt[k] * We[k * H + j];
        float m = xl[(size_t)s_ * H + j] + xr[(size_t)d_ * H + j] + ee;
        m = (m > 0.f) ? m : 0.2f * m;                 // leaky_relu 0.2
        float p = wave_sum(m * att[hH * 32 + lane]);
        if (lane == 0) {
            score[(size_t)e * 4 + hH] = p;
            atomic_max_f32(&maxb[(size_t)d_ * 4 + hH], p);
        }
    }
}
__global__ void gat_edge_exp_k(float* score, const int* dst, int E, int n, const float* maxb, float* den) {
    long long t = blockIdx.x * (long long)blockDim.x + threadIdx.x;
    int ET = E + n;
    if (t >= (long long)ET * 4) return;
    int e = (int)(t >> 2), hH = (int)(t & 3);
    int d_ = (e < E) ? dst[e] : (e - E);
    float v = __expf(score[t] - maxb[(size_t)d_ * 4 + hH]);
    score[t] = v;
    atomicAdd(&den[(size_t)d_ * 4 + hH], v);
}
__global__ void gat_edge_msg_k(const float* xl, const float* score, const float* den,
                               const int* src, const int* dst, int E, int n, float* outb)
{
    int e = blockIdx.x * (blockDim.x >> 5) + (threadIdx.x >> 5);
    int lane = threadIdx.x & 31;
    int ET = E + n;
    if (e >= ET) return;
    int s_ = (e < E) ? src[e] : (e - E);
    int d_ = (e < E) ? dst[e] : (e - E);
#pragma unroll
    for (int hH = 0; hH < 4; ++hH) {
        float alpha = score[(size_t)e * 4 + hH] / (den[(size_t)d_ * 4 + hH] + 1e-16f);
        int j = hH * 32 + lane;
        atomicAdd(&outb[(size_t)d_ * H + j], xl[(size_t)s_ * H + j] * alpha);
    }
}

// ---------------- positional encoding ----------------
__global__ void add_pe_k(float* x, const int* batch, const int* offv, long long tot) {
    long long t = blockIdx.x * (long long)blockDim.x + threadIdx.x;
    if (t >= tot) return;
    int i = (int)(t >> 7), c = (int)(t & (H - 1));
    int pos = i - offv[batch[i]];
    float div = __expf(-(float)(c & ~1) * (9.21034037198f / 128.f));  // ln(10000)
    float ang = (float)pos * div;
    x[t] += (c & 1) ? __cosf(ang) : __sinf(ang);
}

// ---------------- masked cross attention (online softmax; torch-reshape K/V gather) ----------------
__global__ void cross_attn_k(const float* q, const float* kv, int nkv_total,
                             const int* off_q, const float* cnt_q,
                             const int* off_k, const float* cnt_k, float* outb)
{
    int g = blockIdx.x;
    int q0 = off_q[g]; int nq = (int)cnt_q[g];
    int k0 = off_k[g]; int nk = (int)cnt_k[g];
    int half = nkv_total >> 1;
    for (int pair = threadIdx.x; pair < nq * 4; pair += blockDim.x) {
        int ql = pair >> 2, hH = pair & 3;
        int qrow = q0 + ql;
        float qv[32];
#pragma unroll
        for (int c = 0; c < 32; ++c) qv[c] = q[(size_t)qrow * H + hH * 32 + c];
        float m = -3.0e38f, den = 0.f, acc[32];
#pragma unroll
        for (int c = 0; c < 32; ++c) acc[c] = 0.f;
        for (int kk = 0; kk < nk; ++kk) {
            int krow = k0 + kk;  // global key index in reshape(2,n,128) semantics
            const float* kptr = kv + (size_t)(krow >> 1) * 256 + (size_t)(krow & 1) * 128 + hH * 32;
            const float* vptr = kv + (size_t)(half + (krow >> 1)) * 256 + (size_t)(krow & 1) * 128 + hH * 32;
            float s = 0.f;
#pragma unroll
            for (int c = 0; c < 32; ++c) s += qv[c] * kptr[c];
            s *= 0.176776695297f;                       // 1/sqrt(32)
            float mn = fmaxf(m, s);
            float scale = __expf(m - mn);
            float p = __expf(s - mn);
#pragma unroll
            for (int c = 0; c < 32; ++c) acc[c] = acc[c] * scale + p * vptr[c];
            den = den * scale + p;
            m = mn;
        }
        float inv = 1.f / (den + 1e-30f);
#pragma unroll
        for (int c = 0; c < 32; ++c) outb[(size_t)qrow * H + hH * 32 + c] = acc[c] * inv;
    }
}

// ---------------- classifier head ----------------
__global__ void classifier_k(const float* sp, const float* a1p, const float* a2p,
                             const float* w1, const float* b1, const float* w2, const float* b2,
                             float* outp, int n)
{
    int g = blockIdx.x * blockDim.x + threadIdx.x;
    if (g >= n) return;
    float hbuf[8];
    for (int o = 0; o < 8; ++o) {
        float s = b1[o];
        for (int c = 0; c < H; ++c) s += sp[(size_t)g * H + c]  * w1[c * 8 + o];
        for (int c = 0; c < H; ++c) s += a1p[(size_t)g * H + c] * w1[(H + c) * 8 + o];
        for (int c = 0; c < H; ++c) s += a2p[(size_t)g * H + c] * w1[(2 * H + c) * 8 + o];
        hbuf[o] = (s > 0.f) ? s : 0.01f * s;            // leaky_relu 0.01
    }
    float r = b2[0];
    for (int o = 0; o < 8; ++o) r += hbuf[o] * w2[o];
    outp[g] = r;
}

// ==================== host-side helpers ====================
static void gemm(const float* A, const float* W, const float* bias, float* C,
                 int N, int K, int O, int act, hipStream_t s)
{
    int tiles = (N >> 4) * (O >> 5);       // all call sites: N%16==0, O%32==0
    int blocks = (tiles + 7) / 8;
    if ((K & 31) == 0) {
        if (act == ACT_RELU)      gemm_wmma_fast_k<ACT_RELU><<<blocks, 256, 0, s>>>(A, W, bias, C, N, K, O);
        else if (act == ACT_GELU) gemm_wmma_fast_k<ACT_GELU><<<blocks, 256, 0, s>>>(A, W, bias, C, N, K, O);
        else                      gemm_wmma_fast_k<ACT_NONE><<<blocks, 256, 0, s>>>(A, W, bias, C, N, K, O);
    } else {
        if (act == ACT_RELU)      gemm_wmma_pad_k<ACT_RELU><<<blocks, 256, 0, s>>>(A, W, bias, C, N, K, O);
        else if (act == ACT_GELU) gemm_wmma_pad_k<ACT_GELU><<<blocks, 256, 0, s>>>(A, W, bias, C, N, K, O);
        else                      gemm_wmma_pad_k<ACT_NONE><<<blocks, 256, 0, s>>>(A, W, bias, C, N, K, O);
    }
}
static void fillf(float* p, float v, long long n, hipStream_t s) {
    fill_f32_k<<<cdiv64(n, 256), 256, 0, s>>>(p, v, n);
}
static void bn_fw(const float* x, const float* g, const float* b, float* y, int n,
                  float* mu, float* var, hipStream_t s)
{
    col_meanvar_k<<<H, 256, 0, s>>>(x, n, mu, var);
    long long tot = (long long)n * H;
    bn_apply_k<<<cdiv64(tot, 256), 256, 0, s>>>(x, mu, var, g, b, y, tot);
}

// GAT leaves: 0=We 1=att 2=bias 3=bn1_g 4=bn1_b 5=encbn_g 6=encbn_b 7=enc_w 8=enc_b 9=ll_w 10=ll_b 11=lr_w 12=lr_b
static void gat_conv(const float* x, const int* esrc, const int* edst, const float* ea,
                     int E, int n, int edim, const float* const* Lv, float* outb,
                     float* t0, float* t1, float* t2, float* t3,
                     float* la, float* lacnt, float* sc, float* maxb, float* den,
                     float* mu, float* var, hipStream_t s)
{
    long long nH = (long long)n * H;
    int ET = E + n;
    bn_fw(x, Lv[3], Lv[4], t0, n, mu, var, s);
    gemm(t0, Lv[9],  Lv[10], t1, n, H, H, ACT_NONE, s);   // xl
    gemm(t0, Lv[11], Lv[12], t2, n, H, H, ACT_NONE, s);   // xr
    fillf(la, 0.f, (long long)n * edim, s);
    fillf(lacnt, 0.f, n, s);
    seg_accum_ea_k<<<cdiv64((long long)E * edim, 256), 256, 0, s>>>(ea, edst, E, edim, la, lacnt);
    seg_div_k<<<cdiv64((long long)n * edim, 256), 256, 0, s>>>(la, lacnt, n, edim);
    fillf(maxb, -3.0e38f, (long long)n * 4, s);
    fillf(den, 0.f, (long long)n * 4, s);
    fillf(t0, 0.f, nH, s);
    gat_edge_score_k<<<cdiv64(ET, 8), 256, 0, s>>>(t1, t2, ea, la, Lv[0], Lv[1], esrc, edst, E, n, edim, sc, maxb);
    gat_edge_exp_k<<<cdiv64((long long)ET * 4, 256), 256, 0, s>>>(sc, edst, E, n, maxb, den);
    gat_edge_msg_k<<<cdiv64(ET, 8), 256, 0, s>>>(t1, sc, den, esrc, edst, E, n, t0);
    add3_bias_k<<<cdiv64(nH, 256), 256, 0, s>>>(x, t0, Lv[2], t3, nH);   // z = x + gat + bias
    bn_fw(t3, Lv[5], Lv[6], t0, n, mu, var, s);
    gemm(t0, Lv[7], Lv[8], t1, n, H, H, ACT_GELU, s);
    add2_k<<<cdiv64(nH, 256), 256, 0, s>>>(t3, t1, outb, nH);
}

// GIN leaves: 0=encbn_g 1=encbn_b 2=enc_w 3=enc_b 4=eps 5=l1_w 6=l1_b 7=l2_w 8=l2_b
static void gin_conv(const float* xsrc, const float* xdst, const int* esrc, const int* edst, int E,
                     int n, const float* const* Lv, float* outb,
                     float* t0, float* t1, float* t2, float* mu, float* var, hipStream_t s)
{
    long long nH = (long long)n * H;
    fillf(t0, 0.f, nH, s);
    seg_sum_rows_k<<<cdiv64((long long)E * H, 256), 256, 0, s>>>(xsrc, esrc, edst, E, t0);
    gin_combine_k<<<cdiv64(nH, 256), 256, 0, s>>>(xdst, t0, Lv[4], t1, nH);
    gemm(t1, Lv[5], Lv[6], t2, n, H, H, ACT_RELU, s);
    gemm(t2, Lv[7], Lv[8], t0, n, H, H, ACT_NONE, s);
    add2_k<<<cdiv64(nH, 256), 256, 0, s>>>(xdst, t0, t1, nH);            // z
    bn_fw(t1, Lv[0], Lv[1], t0, n, mu, var, s);
    gemm(t0, Lv[2], Lv[3], t2, n, H, H, ACT_GELU, s);
    add2_k<<<cdiv64(nH, 256), 256, 0, s>>>(t1, t2, outb, nH);
}

extern "C" void kernel_launch(void* const* d_in, const int* in_sizes, int n_in,
                              void* d_out, int out_size, void* d_ws, size_t ws_size,
                              hipStream_t stream)
{
    (void)in_sizes; (void)n_in; (void)out_size; (void)ws_size;
    const int NP = 16000, ND = 800, NS = 32, EPP = 320000, EDD = 4800;

    const float* x_prot = (const float*)d_in[0];
    const float* x_drug = (const float*)d_in[1];
    const float* x_super= (const float*)d_in[2];
    const float* pp_ea  = (const float*)d_in[3];
    const float* dd_ea  = (const float*)d_in[4];
    auto PF = [&](int i) -> const float* { return (const float*)d_in[5 + i]; };
    const int* pp_ei = (const int*)d_in[281];
    const int* dd_ei = (const int*)d_in[282];
    const int* ps_ei = (const int*)d_in[283];
    const int* ds_ei = (const int*)d_in[284];
    const int* sp_ei = (const int*)d_in[285];
    const int* sd_ei = (const int*)d_in[286];
    const int* bp = (const int*)d_in[287];
    const int* bd = (const int*)d_in[288];
    const int* bs = (const int*)d_in[289];
    float* outp = (float*)d_out;

    // ---- workspace carve ----
    char* base = (char*)d_ws; size_t woff = 0;
    auto alloc = [&](size_t nelem) -> float* {
        float* p = (float*)(base + woff);
        woff += ((nelem * sizeof(float)) + 255) & ~(size_t)255;
        return p;
    };
    float* XP = alloc((size_t)NP * H);
    float* P0 = alloc((size_t)NP * H); float* P1 = alloc((size_t)NP * H);
    float* P2 = alloc((size_t)NP * H); float* P3 = alloc((size_t)NP * H);
    float* P4 = alloc((size_t)NP * H);
    float* XD = alloc((size_t)ND * H);
    float* D0 = alloc((size_t)ND * H); float* D1 = alloc((size_t)ND * H);
    float* D2 = alloc((size_t)ND * H); float* D3 = alloc((size_t)ND * H);
    float* D4 = alloc((size_t)ND * H);
    float* XS = alloc((size_t)NS * H);
    float* S0 = alloc((size_t)NS * H); float* S1 = alloc((size_t)NS * H);
    float* S2 = alloc((size_t)NS * H); float* S3 = alloc((size_t)NS * H);
    float* S4 = alloc((size_t)NS * H);
    float* KV   = alloc((size_t)NP * 256);
    float* SC   = alloc((size_t)(EPP + NP) * 4);
    float* MAXB = alloc((size_t)NP * 4);
    float* DEN  = alloc((size_t)NP * 4);
    float* LA   = alloc((size_t)NP * 8);
    float* LACNT= alloc(NP);
    float* MU = alloc(H); float* VAR = alloc(H);
    float* CP = alloc(NS); float* CD = alloc(NS); float* CS = alloc(NS);
    int* OP = (int*)alloc(NS); int* OD = (int*)alloc(NS); int* OS = (int*)alloc(NS);

    // ---- input projections (ragged K -> padded path) ----
    gemm(x_prot,  PF(272), PF(273), XP, NP, 41, H, ACT_NONE, stream);
    gemm(x_drug,  PF(270), PF(271), XD, ND, 44, H, ACT_NONE, stream);
    gemm(x_super, PF(274), PF(275), XS, NS, 64, H, ACT_NONE, stream);

    // ---- 4 hetero layers ----
    for (int L = 0; L < 4; ++L) {
        int LB = 22 + L * 62;   // layers[L], sorted keys: dd(13) ds(9) pp(13) ps(9) sd(9) sp(9)
        const float* Gdd[13]; const float* Gpp[13];
        const float* Ids[9]; const float* Ips[9]; const float* Isd[9]; const float* Isp[9];
        for (int i = 0; i < 13; ++i) { Gdd[i] = PF(LB + i); Gpp[i] = PF(LB + 22 + i); }
        for (int i = 0; i < 9;  ++i) {
            Ids[i] = PF(LB + 13 + i); Ips[i] = PF(LB + 35 + i);
            Isd[i] = PF(LB + 44 + i); Isp[i] = PF(LB + 53 + i);
        }
        // protein: egret(pp) and gin(s->p), mean
        gat_conv(XP, pp_ei, pp_ei + EPP, pp_ea, EPP, NP, 8, Gpp, P4,
                 P0, P1, P2, P3, LA, LACNT, SC, MAXB, DEN, MU, VAR, stream);
        gin_conv(XS, XP, sp_ei, sp_ei + NP, NP, NP, Isp, P3, P0, P1, P2, MU, VAR, stream);
        avg2_k<<<cdiv64((long long)NP * H, 256), 256, 0, stream>>>(P4, P3, P0, (long long)NP * H);
        // drug
        gat_conv(XD, dd_ei, dd_ei + EDD, dd_ea, EDD, ND, 6, Gdd, D4,
                 D0, D1, D2, D3, LA, LACNT, SC, MAXB, DEN, MU, VAR, stream);
        gin_conv(XS, XD, sd_ei, sd_ei + ND, ND, ND, Isd, D3, D0, D1, D2, MU, VAR, stream);
        avg2_k<<<cdiv64((long long)ND * H, 256), 256, 0, stream>>>(D4, D3, D0, (long long)ND * H);
        // super (uses old XP/XD, still intact)
        gin_conv(XP, XS, ps_ei, ps_ei + NP, NP, NS, Ips, S3, S0, S1, S2, MU, VAR, stream);
        gin_conv(XD, XS, ds_ei, ds_ei + ND, ND, NS, Ids, S4, S0, S1, S2, MU, VAR, stream);
        avg2_k<<<cdiv64((long long)NS * H, 256), 256, 0, stream>>>(S3, S4, S0, (long long)NS * H);
        // shared post batch-norms: d=12,13  p=14,15  s=16,17
        bn_fw(P0, PF(14), PF(15), XP, NP, MU, VAR, stream);
        bn_fw(D0, PF(12), PF(13), XD, ND, MU, VAR, stream);
        bn_fw(S0, PF(16), PF(17), XS, NS, MU, VAR, stream);
    }

    // ---- batch offsets / counts ----
    fill_i32_k<<<1, 64, 0, stream>>>(OP, 0x7FFFFFFF, NS);
    fill_i32_k<<<1, 64, 0, stream>>>(OD, 0x7FFFFFFF, NS);
    fill_i32_k<<<1, 64, 0, stream>>>(OS, 0x7FFFFFFF, NS);
    fillf(CP, 0.f, NS, stream); fillf(CD, 0.f, NS, stream); fillf(CS, 0.f, NS, stream);
    batch_offsets_k<<<cdiv64(NP, 256), 256, 0, stream>>>(bp, NP, OP, CP);
    batch_offsets_k<<<cdiv64(ND, 256), 256, 0, stream>>>(bd, ND, OD, CD);
    batch_offsets_k<<<cdiv64(NS, 256), 256, 0, stream>>>(bs, NS, OS, CS);

    // ---- positional encodings ----
    add_pe_k<<<cdiv64((long long)NP * H, 256), 256, 0, stream>>>(XP, bp, OP, (long long)NP * H);
    add_pe_k<<<cdiv64((long long)ND * H, 256), 256, 0, stream>>>(XD, bd, OD, (long long)ND * H);

    // ---- cross attention 1: protein queries, drug K/V ----
    gemm(XP, PF(0), PF(1), P0, NP, H, H,   ACT_NONE, stream);   // q1
    gemm(XD, PF(2), PF(3), KV, ND, H, 256, ACT_NONE, stream);   // kv1
    cross_attn_k<<<NS, 128, 0, stream>>>(P0, KV, ND, OP, CP, OD, CD, P1);
    gemm(P1, PF(4), PF(5), P2, NP, H, H, ACT_NONE, stream);     // a1
    // ---- cross attention 2: drug queries, protein K/V ----
    gemm(XD, PF(6), PF(7), D0, ND, H, H,   ACT_NONE, stream);
    gemm(XP, PF(8), PF(9), KV, NP, H, 256, ACT_NONE, stream);
    cross_attn_k<<<NS, 128, 0, stream>>>(D0, KV, NP, OD, CD, OP, CP, D1);
    gemm(D1, PF(10), PF(11), D2, ND, H, H, ACT_NONE, stream);   // a2

    // ---- pooled means ----
    fillf(S0, 0.f, (long long)NS * H, stream);
    fillf(S1, 0.f, (long long)NS * H, stream);
    fillf(S2, 0.f, (long long)NS * H, stream);
    seg_mean_pool_k<<<cdiv64((long long)NS * H, 256), 256, 0, stream>>>(XS, bs, CS, (long long)NS * H, S0);
    seg_mean_pool_k<<<cdiv64((long long)NP * H, 256), 256, 0, stream>>>(P2, bp, CP, (long long)NP * H, S1);
    seg_mean_pool_k<<<cdiv64((long long)ND * H, 256), 256, 0, stream>>>(D2, bd, CD, (long long)ND * H, S2);

    // ---- classifier head -> d_out[32] ----
    classifier_k<<<1, 64, 0, stream>>>(S0, S1, S2, PF(18), PF(19), PF(20), PF(21), outp, NS);
}
